// LstmAutoencoderPk_77309412086
// MI455X (gfx1250) — compile-verified, matrix operand order checked by reading, not run on hardware
//
#include <hip/hip_runtime.h>
#include <hip/hip_bf16.h>

// ---------------------------------------------------------------------------
// Types for CDNA5 WMMA (wave32): v_wmma_f32_16x16x32_bf16
// ---------------------------------------------------------------------------
typedef __bf16 bf16_t;
typedef bf16_t v16bf __attribute__((ext_vector_type(16)));
typedef bf16_t v8bf  __attribute__((ext_vector_type(8)));
typedef float  v8f   __attribute__((ext_vector_type(8)));
typedef int    v4i   __attribute__((ext_vector_type(4)));

#define B_DIM 64
#define T_DIM 4096
#define I_DIM 128
#define H_DIM 256
#define G4    1024         // 4*H
#define K_DIM 384          // I + H (concatenated contraction)
#define H_PITCH 264        // 256 + 8 pad: 4-bank stride in LDS -> conflict-free b128
#define X_PITCH 136        // 128 + 8 pad: 68 DW stride (4 banks) -> conflict-free b128

// ---------------------------------------------------------------------------
// CDNA5 async global->LDS path (ASYNCcnt), guarded so the kernel always
// compiles: if the builtin is absent we fall back to direct global loads.
// Builtin signature (from hipcc diagnostic): (AS1 int4*, AS3 int4*, imm, imm).
// ---------------------------------------------------------------------------
#if defined(__has_builtin)
#if __has_builtin(__builtin_amdgcn_global_load_async_to_lds_b128)
#define HAVE_ASYNC_LDS 1
#endif
#endif

#ifdef HAVE_ASYNC_LDS
typedef __attribute__((address_space(1))) v4i gv4i_t;
typedef __attribute__((address_space(3))) v4i lv4i_t;
__device__ __forceinline__ void async_b128(const bf16_t* g, bf16_t* l) {
    __builtin_amdgcn_global_load_async_to_lds_b128((gv4i_t*)g, (lv4i_t*)l, 0, 0);
}
__device__ __forceinline__ void wait_async0() {
#if __has_builtin(__builtin_amdgcn_s_wait_asynccnt)
    __builtin_amdgcn_s_wait_asynccnt(0);
#else
    asm volatile("s_wait_asynccnt 0x0" ::: "memory");
#endif
}
#endif

// ---------------------------------------------------------------------------
// Fragment load per ISA 16-bit A/B layout:
//   lanes 0-15  hold K = {kc..kc+7, kc+16..kc+23}
//   lanes 16-31 hold K = {kc+8..kc+15, kc+24..kc+31}
// i.e. two contiguous 16B chunks at (kc + 8*khalf) and (+16).
// ---------------------------------------------------------------------------
__device__ __forceinline__ v16bf load_frag(const bf16_t* rowp, int kc, int khalf) {
    const bf16_t* p = rowp + kc + khalf * 8;
    v8bf lo = *(const v8bf*)(p);
    v8bf hi = *(const v8bf*)(p + 16);
    return __builtin_shufflevector(lo, hi, 0,1,2,3,4,5,6,7,8,9,10,11,12,13,14,15);
}

__device__ __forceinline__ float sigm_f(float x) { return 1.0f / (1.0f + __expf(-x)); }
__device__ __forceinline__ float tanh_f(float x) { return 2.0f / (1.0f + __expf(-2.0f * x)) - 1.0f; }

// ---------------------------------------------------------------------------
// Prep kernels
// ---------------------------------------------------------------------------
__global__ void k_cvt_bf16(const float* __restrict__ src, bf16_t* __restrict__ dst, size_t n) {
    size_t i = (size_t)blockIdx.x * blockDim.x + threadIdx.x;
    size_t stride = (size_t)gridDim.x * blockDim.x;
    for (; i < n; i += stride) dst[i] = (bf16_t)src[i];
}

// Wcat[slot][n][k] : k<128 -> Wih[n][k], else Whh[n][k-128]. 3 slots (fwd, bwd, dec).
__global__ void k_pack_w(const float* __restrict__ wih0, const float* __restrict__ whh0,
                         const float* __restrict__ wih1, const float* __restrict__ whh1,
                         const float* __restrict__ wih2, const float* __restrict__ whh2,
                         bf16_t* __restrict__ wcat) {
    size_t n_tot = (size_t)3 * G4 * K_DIM;
    size_t i = (size_t)blockIdx.x * blockDim.x + threadIdx.x;
    size_t stride = (size_t)gridDim.x * blockDim.x;
    for (; i < n_tot; i += stride) {
        int slot = (int)(i / ((size_t)G4 * K_DIM));
        int rem  = (int)(i % ((size_t)G4 * K_DIM));
        int n = rem / K_DIM, k = rem % K_DIM;
        const float* wih = (slot == 0) ? wih0 : (slot == 1) ? wih1 : wih2;
        const float* whh = (slot == 0) ? whh0 : (slot == 1) ? whh1 : whh2;
        float v = (k < I_DIM) ? wih[(size_t)n * I_DIM + k]
                              : whh[(size_t)n * H_DIM + (k - I_DIM)];
        wcat[i] = (bf16_t)v;
    }
}

__global__ void k_pack_bias(const float* __restrict__ bih0, const float* __restrict__ bhh0,
                            const float* __restrict__ bih1, const float* __restrict__ bhh1,
                            const float* __restrict__ bih2, const float* __restrict__ bhh2,
                            float* __restrict__ bias) {
    int i = blockIdx.x * blockDim.x + threadIdx.x;
    if (i >= 3 * G4) return;
    int slot = i / G4, n = i % G4;
    const float* bih = (slot == 0) ? bih0 : (slot == 1) ? bih1 : bih2;
    const float* bhh = (slot == 0) ? bhh0 : (slot == 1) ? bhh1 : bhh2;
    bias[i] = bih[n] + bhh[n];
}

// ---------------------------------------------------------------------------
// Persistent recurrent LSTM kernel.
// 512 threads = 16 waves; wave w owns nh-tile w (16 of 256 cols) and ALL
// 4 m-tiles x 4 gates -> 16 independent accumulator tiles (no WMMA RAW nops).
//  - weight fragments loaded exactly once per block per step, reused 4x
//  - i,f,g,o for the same (m,nh) stay in-wave; c lives in VGPRs across all T
//  - x_t double-buffered in LDS via GLOBAL_LOAD_ASYNC_TO_LDS_B128, issued one
//    step ahead (overlaps the whole step's compute); fallback = global loads
//  - h (64x256 bf16) exchanged via LDS, 2 barriers per step
// ---------------------------------------------------------------------------
template <bool WRITE_HS>
__global__ __launch_bounds__(512, 1)
void k_lstm(const bf16_t* __restrict__ x_bf,      // [B, T_DIM, I]
            const bf16_t* __restrict__ wcat_base, // [3][G4][K_DIM]
            const float*  __restrict__ bias_base, // [3][G4]
            int slot_base, int rev_mask, int steps,
            const float*  __restrict__ h_init,    // [B,H] or null (zeros)
            bf16_t*       __restrict__ hs_out,    // [B, steps, H] or null
            float*        __restrict__ hT_base)   // [nblocks][B,H] or null
{
    __shared__ bf16_t h_lds[B_DIM * H_PITCH];
#ifdef HAVE_ASYNC_LDS
    __shared__ bf16_t x_stage[2 * B_DIM * X_PITCH];
#endif

    const int tid    = threadIdx.x;
    const int wave   = tid >> 5;          // 0..15 == nh tile
    const int lane   = tid & 31;
    const int lane15 = lane & 15;
    const int khalf  = lane >> 4;

    const int slot = slot_base + blockIdx.x;
    const bf16_t* Wcat = wcat_base + (size_t)slot * G4 * K_DIM;
    const float*  bias = bias_base + (size_t)slot * G4;
    const int rev = (rev_mask >> blockIdx.x) & 1;
    const int t0  = rev ? (steps - 1) : 0;
    const int dt  = rev ? -1 : 1;

    // init h in LDS
    for (int idx = tid; idx < B_DIM * H_DIM; idx += 512) {
        int r = idx >> 8, c = idx & 255;
        float v = h_init ? h_init[idx] : 0.0f;
        h_lds[r * H_PITCH + c] = (bf16_t)v;
    }

    // hoisted row pointers (time-invariant)
    const bf16_t* wrow[4];   // weight row base per gate
    float bias_g[4];
#pragma unroll
    for (int g = 0; g < 4; ++g) {
        const int nrow = g * H_DIM + wave * 16 + lane15;
        wrow[g]   = Wcat + (size_t)nrow * K_DIM;
        bias_g[g] = bias[nrow];
    }
    const bf16_t* hrow[4];   // LDS row base per m-tile
#pragma unroll
    for (int mt = 0; mt < 4; ++mt)
        hrow[mt] = &h_lds[(mt * 16 + lane15) * H_PITCH];
    const int n_col = wave * 16 + lane15;

#ifdef HAVE_ASYNC_LDS
    // prologue: async-stage x_{t0} into buffer 0 (1024 b128 chunks / 512 thr)
    {
        const size_t xoff = (size_t)t0 * I_DIM;
#pragma unroll
        for (int it = 0; it < 2; ++it) {
            int c = tid + it * 512;
            int row = c >> 4, ch = c & 15;
            async_b128(x_bf + (size_t)row * T_DIM * I_DIM + xoff + ch * 8,
                       &x_stage[row * X_PITCH + ch * 8]);
        }
        wait_async0();
    }
    int buf = 0;
#endif
    __syncthreads();

    // persistent cell state: 4 m-tiles x 8 rows
    float creg[4][8];
#pragma unroll
    for (int mt = 0; mt < 4; ++mt)
#pragma unroll
        for (int j = 0; j < 8; ++j) creg[mt][j] = 0.0f;

    for (int t = 0; t < steps; ++t) {
#ifdef HAVE_ASYNC_LDS
        // issue async stage of x_{t+1} into the other buffer; overlaps compute
        if (t + 1 < steps) {
            const size_t xoff = (size_t)(t0 + dt * (t + 1)) * I_DIM;
            bf16_t* st = &x_stage[(buf ^ 1) * B_DIM * X_PITCH];
#pragma unroll
            for (int it = 0; it < 2; ++it) {
                int c = tid + it * 512;
                int row = c >> 4, ch = c & 15;
                async_b128(x_bf + (size_t)row * T_DIM * I_DIM + xoff + ch * 8,
                           st + row * X_PITCH + ch * 8);
            }
        }
        const bf16_t* xs = &x_stage[buf * B_DIM * X_PITCH];
#else
        const size_t xoff = (size_t)(t0 + dt * t) * I_DIM;
#endif

        v8f acc[4][4];
#pragma unroll
        for (int mt = 0; mt < 4; ++mt)
#pragma unroll
            for (int g = 0; g < 4; ++g)
#pragma unroll
                for (int j = 0; j < 8; ++j) acc[mt][g][j] = bias_g[g];

        // ---- x phase: K chunks 0..3 ----
#pragma unroll
        for (int kc = 0; kc < 4; ++kc) {
            v16bf a[4], b[4];
#pragma unroll
            for (int mt = 0; mt < 4; ++mt) {
#ifdef HAVE_ASYNC_LDS
                a[mt] = load_frag(xs + (mt * 16 + lane15) * X_PITCH, kc * 32, khalf);
#else
                a[mt] = load_frag(x_bf + (size_t)(mt * 16 + lane15) * T_DIM * I_DIM + xoff,
                                  kc * 32, khalf);
#endif
            }
#pragma unroll
            for (int g = 0; g < 4; ++g) b[g] = load_frag(wrow[g], kc * 32, khalf);
#pragma unroll
            for (int g = 0; g < 4; ++g)
#pragma unroll
                for (int mt = 0; mt < 4; ++mt)
                    acc[mt][g] = __builtin_amdgcn_wmma_f32_16x16x32_bf16(
                        false, a[mt], false, b[g], (short)0, acc[mt][g], false, false);
        }

        // ---- h phase: K chunks 0..7, A from LDS (DScnt) + B from L2 (LOADcnt) ----
#pragma unroll
        for (int kc = 0; kc < 8; ++kc) {
            v16bf a[4], b[4];
#pragma unroll
            for (int mt = 0; mt < 4; ++mt) a[mt] = load_frag(hrow[mt], kc * 32, khalf);
#pragma unroll
            for (int g = 0; g < 4; ++g)    b[g]  = load_frag(wrow[g], I_DIM + kc * 32, khalf);
#pragma unroll
            for (int g = 0; g < 4; ++g)
#pragma unroll
                for (int mt = 0; mt < 4; ++mt)
                    acc[mt][g] = __builtin_amdgcn_wmma_f32_16x16x32_bf16(
                        false, a[mt], false, b[g], (short)0, acc[mt][g], false, false);
        }

        __syncthreads();   // all h reads for step t are done

        // gate combine (i,f,g,o all in-wave) + write new h
#pragma unroll
        for (int mt = 0; mt < 4; ++mt) {
            const int mbase = mt * 16 + 8 * khalf;   // D: lanes16-31 -> M=j+8
#pragma unroll
            for (int j = 0; j < 8; ++j) {
                float ip = acc[mt][0][j], fp = acc[mt][1][j];
                float gp = acc[mt][2][j], op = acc[mt][3][j];
                float c = sigm_f(fp) * creg[mt][j] + sigm_f(ip) * tanh_f(gp);
                creg[mt][j] = c;
                float h = sigm_f(op) * tanh_f(c);
                h_lds[(mbase + j) * H_PITCH + n_col] = (bf16_t)h;
            }
        }

#ifdef HAVE_ASYNC_LDS
        wait_async0();     // own async issues done; barrier makes all visible
        buf ^= 1;
#endif
        __syncthreads();   // new h (and staged x) visible to all waves

        if (WRITE_HS) {
            // cooperative vectorized hs store: 2048 b128 chunks / 512 threads
#pragma unroll
            for (int it = 0; it < 4; ++it) {
                int c = tid + it * 512;
                int row = c >> 5, ch = c & 31;
                v8bf v = *(const v8bf*)&h_lds[row * H_PITCH + ch * 8];
                *(v8bf*)&hs_out[((size_t)row * steps + t) * H_DIM + ch * 8] = v;
            }
        }
    }

    if (hT_base) {
        float* hT = hT_base + (size_t)blockIdx.x * B_DIM * H_DIM;
        for (int idx = tid; idx < B_DIM * H_DIM; idx += 512) {
            int r = idx >> 8, c = idx & 255;
            hT[idx] = (float)h_lds[r * H_PITCH + c];
        }
    }
}

// ---------------------------------------------------------------------------
// latent = concat(hT_f, hT_b) @ lat_W^T + lat_b ; hidden = latent @ l2h_W^T + l2h_b
// Tiny (3 MFLOP) -> one block, scalar fp32.
// ---------------------------------------------------------------------------
__global__ __launch_bounds__(256)
void k_latent(const float* __restrict__ hTf, const float* __restrict__ hTb,
              const float* __restrict__ latW, const float* __restrict__ latb,
              const float* __restrict__ l2hW, const float* __restrict__ l2hb,
              float* __restrict__ hidden) {
    __shared__ float lat[64 * 64];
    for (int idx = threadIdx.x; idx < 64 * 64; idx += 256) {
        int b = idx >> 6, l = idx & 63;
        float s = latb[l];
        const float* wr = latW + (size_t)l * 512;
        for (int j = 0; j < 256; ++j) s += hTf[b * 256 + j] * wr[j];
        for (int j = 0; j < 256; ++j) s += hTb[b * 256 + j] * wr[256 + j];
        lat[idx] = s;
    }
    __syncthreads();
    for (int idx = threadIdx.x; idx < 64 * 256; idx += 256) {
        int b = idx >> 8, h = idx & 255;
        float s = l2hb[h];
        const float* wr = l2hW + (size_t)h * 64;
        for (int l = 0; l < 64; ++l) s += lat[b * 64 + l] * wr[l];
        hidden[idx] = s;
    }
}

// ---------------------------------------------------------------------------
// decoded = hs @ out_W^T + out_b : [262080,256] x [128,256]^T -> [262080,128]
// Block = 8 waves covering 64 rows x 128 cols: wave w owns n-tile w and
// 4 independent m-tile accumulators -> WMMA RAW hazard slots filled with
// useful (independent) WMMAs; each B fragment reused 4x.
// ---------------------------------------------------------------------------
__global__ __launch_bounds__(256)
void k_outproj(const bf16_t* __restrict__ hs,   // [B*(T-1), H]
               const bf16_t* __restrict__ wo,   // [I, H] bf16
               const float*  __restrict__ outb, // [I]
               float* __restrict__ out)         // [B*(T-1), I]
{
    const int wave = threadIdx.x >> 5;
    const int lane = threadIdx.x & 31;
    const int lane15 = lane & 15;
    const int khalf  = lane >> 4;
    const int m0 = blockIdx.x * 64;
    const int n  = wave * 16 + lane15;

    v8f acc[4];
#pragma unroll
    for (int mt = 0; mt < 4; ++mt)
#pragma unroll
        for (int j = 0; j < 8; ++j) acc[mt][j] = outb[n];

    const bf16_t* arow[4];
#pragma unroll
    for (int mt = 0; mt < 4; ++mt)
        arow[mt] = hs + (size_t)(m0 + mt * 16 + lane15) * H_DIM;
    const bf16_t* brow = wo + (size_t)n * H_DIM;

#pragma unroll
    for (int kc = 0; kc < 8; ++kc) {
        v16bf b = load_frag(brow, kc * 32, khalf);
#pragma unroll
        for (int mt = 0; mt < 4; ++mt) {
            v16bf a = load_frag(arow[mt], kc * 32, khalf);
            acc[mt] = __builtin_amdgcn_wmma_f32_16x16x32_bf16(
                false, a, false, b, (short)0, acc[mt], false, false);
        }
    }
#pragma unroll
    for (int mt = 0; mt < 4; ++mt) {
        const int mbase = m0 + mt * 16 + 8 * khalf;
#pragma unroll
        for (int j = 0; j < 8; ++j)
            out[(size_t)(mbase + j) * I_DIM + n] = acc[mt][j];
    }
}

// ---------------------------------------------------------------------------
// Host launcher
// ---------------------------------------------------------------------------
extern "C" void kernel_launch(void* const* d_in, const int* in_sizes, int n_in,
                              void* d_out, int out_size, void* d_ws, size_t ws_size,
                              hipStream_t stream) {
    const float* x        = (const float*)d_in[0];
    const float* wih_f    = (const float*)d_in[1];
    const float* whh_f    = (const float*)d_in[2];
    const float* bih_f    = (const float*)d_in[3];
    const float* bhh_f    = (const float*)d_in[4];
    const float* wih_b    = (const float*)d_in[5];
    const float* whh_b    = (const float*)d_in[6];
    const float* bih_b    = (const float*)d_in[7];
    const float* bhh_b    = (const float*)d_in[8];
    const float* latW     = (const float*)d_in[9];
    const float* latb     = (const float*)d_in[10];
    const float* l2hW     = (const float*)d_in[11];
    const float* l2hb     = (const float*)d_in[12];
    const float* wih_d    = (const float*)d_in[13];
    const float* whh_d    = (const float*)d_in[14];
    const float* bih_d    = (const float*)d_in[15];
    const float* bhh_d    = (const float*)d_in[16];
    const float* outW     = (const float*)d_in[17];
    const float* outb     = (const float*)d_in[18];
    float* out = (float*)d_out;

    // workspace carve-up (256B aligned)
    char* wsb = (char*)d_ws;
    size_t off = 0;
    auto carve = [&](size_t bytes) -> char* {
        char* p = wsb + off;
        off = (off + bytes + 255) & ~(size_t)255;
        return p;
    };
    bf16_t* x_bf   = (bf16_t*)carve((size_t)B_DIM * T_DIM * I_DIM * sizeof(bf16_t));       // 64 MB
    bf16_t* hs_bf  = (bf16_t*)carve((size_t)B_DIM * (T_DIM - 1) * H_DIM * sizeof(bf16_t)); // 128 MB
    bf16_t* wcat   = (bf16_t*)carve((size_t)3 * G4 * K_DIM * sizeof(bf16_t));
    float*  bias   = (float*) carve((size_t)3 * G4 * sizeof(float));
    bf16_t* wo_bf  = (bf16_t*)carve((size_t)I_DIM * H_DIM * sizeof(bf16_t));
    float*  hT     = (float*) carve((size_t)2 * B_DIM * H_DIM * sizeof(float));
    float*  hidden = (float*) carve((size_t)B_DIM * H_DIM * sizeof(float));
    (void)ws_size; (void)in_sizes; (void)n_in; (void)out_size;

    // Stage 0: precision conversion + weight packing
    k_cvt_bf16<<<4096, 256, 0, stream>>>(x, x_bf, (size_t)B_DIM * T_DIM * I_DIM);
    k_cvt_bf16<<<128, 256, 0, stream>>>(outW, wo_bf, (size_t)I_DIM * H_DIM);
    k_pack_w<<<2048, 256, 0, stream>>>(wih_f, whh_f, wih_b, whh_b, wih_d, whh_d, wcat);
    k_pack_bias<<<(3 * G4 + 255) / 256, 256, 0, stream>>>(bih_f, bhh_f, bih_b, bhh_b,
                                                          bih_d, bhh_d, bias);

    // Stage 1: forward + backward encoder LSTMs (2 concurrent workgroups)
    k_lstm<false><<<2, 512, 0, stream>>>(x_bf, wcat, bias,
                                         /*slot_base=*/0, /*rev_mask=*/0b10, /*steps=*/T_DIM,
                                         /*h_init=*/nullptr, /*hs_out=*/nullptr, /*hT=*/hT);

    // Stage 2: latent bottleneck -> decoder initial hidden state
    k_latent<<<1, 256, 0, stream>>>(hT, hT + (size_t)B_DIM * H_DIM,
                                    latW, latb, l2hW, l2hb, hidden);

    // Stage 3: decoder LSTM over T-1 steps, streaming hs to global (bf16)
    k_lstm<true><<<1, 512, 0, stream>>>(x_bf, wcat, bias,
                                        /*slot_base=*/2, /*rev_mask=*/0, /*steps=*/T_DIM - 1,
                                        /*h_init=*/hidden, /*hs_out=*/hs_bf, /*hT=*/nullptr);

    // Stage 4: output projection (chip-wide WMMA GEMM)
    const int m_blocks = (B_DIM * (T_DIM - 1)) / 64; // 262080 / 64 = 4095
    k_outproj<<<m_blocks, 256, 0, stream>>>(hs_bf, wo_bf, outb, out);
}
